// MultiHeadAttention_3710851743971
// MI455X (gfx1250) — matrix operand order repaired and checked by reading, not compiled, not run on hardware
//
#include <hip/hip_runtime.h>
#include <hip/hip_bf16.h>

// ---------------------------------------------------------------------------
// MI455X (gfx1250) multi-head attention
//   bf16 WMMA everywhere + flash-attention streaming + TDM (tensor_load_to_lds)
//   double-buffered K/V tiles in LDS, s_wait_tensorcnt sync.
// B=2, S=2048, D=128, H=8, D_PROJ=1024
// ---------------------------------------------------------------------------

#define B_   2
#define S_   2048
#define D_   128
#define H_   8
#define DP_  1024
#define ROWS (B_ * S_)          // 4096
#define QSCALE 0.12751744654f   // log2(e) / sqrt(128)
#define NKB  (S_ / 64)          // 32 key blocks

typedef __bf16 bf16;
typedef unsigned int u32;
typedef __attribute__((ext_vector_type(8)))  __bf16 bf16x8;
typedef __attribute__((ext_vector_type(16))) __bf16 bf16x16;
typedef __attribute__((ext_vector_type(8)))  float  floatx8;
typedef __attribute__((ext_vector_type(4)))  u32    uint32x4;
typedef __attribute__((ext_vector_type(4)))  int    int32x4;
typedef __attribute__((ext_vector_type(8)))  int    int32x8;

union V16U { bf16x16 v; bf16x8 h[2]; };

__device__ __forceinline__ bf16x16 join16(bf16x8 lo, bf16x8 hi) {
    V16U u; u.h[0] = lo; u.h[1] = hi; return u.v;
}

__device__ __forceinline__ floatx8 wmma_bf16(bf16x16 a, bf16x16 b, floatx8 c) {
    // D = A(16x32 bf16) * B(32x16 bf16) + C(16x16 f32)
    return __builtin_amdgcn_wmma_f32_16x16x32_bf16(
        false, a, false, b, (short)0, c, false, false);
}

// A-fragment (16x32 bf16): lane%16 = row M; 16 contiguous bf16 at
// k = kbase + (lane/16)*16 (two 16B chunks).
__device__ __forceinline__ bf16x16 load_afrag(const bf16* __restrict__ row,
                                              int kbase, int half) {
    const bf16* p = row + kbase + half * 16;
    return join16(*(const bf16x8*)(p), *(const bf16x8*)(p + 8));
}

// B-fragment from a (possibly LDS) row base: lane%16 selects the row pointer
// outside; here we read 16 contiguous bf16 at koff as two 16B chunks.
__device__ __forceinline__ bf16x16 load_bfrag16(const bf16* row, int koff) {
    return join16(*(const bf16x8*)(row + koff), *(const bf16x8*)(row + koff + 8));
}

// ---------------------------------------------------------------------------
// Tensor Data Mover: 2D tile load Global -> LDS (CDNA5 ISA)
// group0: count=1 | lds_addr | global_addr[56:0] | type=2
// group1: data_size=2B, pad cfg, tensor_dim0/1, tile_dim0/1, dim0 stride
__device__ __forceinline__ void tdm_load_2d(u32 lds_addr, const void* gptr,
                                            u32 tensor_d0, u32 tensor_d1,
                                            u32 tile_d0, u32 tile_d1,
                                            u32 stride0, u32 flags_w0) {
    unsigned long long ga = (unsigned long long)(uintptr_t)gptr;
    uint32x4 g0;
    g0[0] = 1u;                                   // count=1 (valid descriptor)
    g0[1] = lds_addr;                             // LDS byte address
    g0[2] = (u32)ga;                              // global_addr[31:0]
    g0[3] = (u32)((ga >> 32) & 0x1FFFFFFu) | (2u << 30);  // addr[56:32] | type=2
    int32x8 g1;
    g1[0] = (int)flags_w0;                                    // data_size|pad cfg
    g1[1] = (int)((tensor_d0 & 0xFFFFu) << 16);               // tensor_dim0 lo
    g1[2] = (int)((tensor_d0 >> 16) | ((tensor_d1 & 0xFFFFu) << 16));
    g1[3] = (int)((tensor_d1 >> 16) | (tile_d0 << 16));       // tile_dim0
    g1[4] = (int)(tile_d1 & 0xFFFFu);                         // tile_dim1, tile_dim2=0
    g1[5] = (int)stride0;                                     // tensor_dim0_stride lo
    g1[6] = 0;
    g1[7] = 0;
    int32x4 z4 = (int32x4)0;
#if defined(__clang_major__) && (__clang_major__ >= 23)
    int32x8 z8 = (int32x8)0;
    __builtin_amdgcn_tensor_load_to_lds(g0, g1, z4, z4, z8, 0);
#else
    __builtin_amdgcn_tensor_load_to_lds(g0, g1, z4, z4, 0);
#endif
}

// data_size=2B (1<<16), pad_enable (1<<20), pad_interval code, pad_amount code.
// K tile rows: 128 bf16 = 64 dwords -> interval code 5; pad 4 dwords -> code 3.
// V tile rows:  64 bf16 = 32 dwords -> interval code 4; pad 4 dwords -> code 3.
#define TDM_FLAGS_K ((1u << 16) | (1u << 20) | (5u << 22) | (3u << 25))
#define TDM_FLAGS_V ((1u << 16) | (1u << 20) | (4u << 22) | (3u << 25))

// ---------------------------------------------------------------------------
// elementwise fp32 -> bf16
__global__ void k_cvt_bf16(const float* __restrict__ src, bf16* __restrict__ dst, int n) {
    int i = blockIdx.x * blockDim.x + threadIdx.x;
    if (i < n) dst[i] = (bf16)src[i];
}

// transpose + convert: src [R,C] fp32 row-major -> dst [C,R] bf16 row-major
__global__ void k_cvtT_bf16(const float* __restrict__ src, bf16* __restrict__ dst,
                            int R, int C) {
    int i = blockIdx.x * blockDim.x + threadIdx.x;
    if (i < R * C) {
        int r = i / C, c = i % C;
        dst[(size_t)c * R + r] = (bf16)src[i];
    }
}

// ---------------------------------------------------------------------------
// Projection GEMM: Y[4096,1024] = X[4096,128](bf16) @ W (bf16, stored W^T) + bias
// MODE 0: Q -> [b,h,s,d], pre-scaled by log2e/sqrt(128), mask folded (0-scale row)
// MODE 1: K -> [b,h,s,d]
// MODE 2: V -> [b,h,d,s]  (transposed so P@V B-fragments are contiguous)
template <int MODE>
__global__ __launch_bounds__(256) void k_proj(const bf16* __restrict__ X,
                                              const bf16* __restrict__ WT,
                                              const float* __restrict__ bias,
                                              const unsigned char* __restrict__ mask,
                                              bf16* __restrict__ out) {
    const int wave = threadIdx.x >> 5;
    const int lane = threadIdx.x & 31;
    const int half = lane >> 4;
    const int l16  = lane & 15;
    const int tile = blockIdx.x * 8 + wave;   // 256 * 64 tiles
    const int mt = tile >> 6;
    const int nt = tile & 63;
    const int row0 = mt * 16;
    const int col0 = nt * 16;

    floatx8 acc = (floatx8)(0.0f);
    const bf16* arow = X + (size_t)(row0 + l16) * D_;
#pragma unroll
    for (int c = 0; c < 4; ++c) {
        bf16x16 a = load_afrag(arow, c * 32, half);
        bf16x16 b = *(const bf16x16*)(WT + (size_t)(col0 + l16) * D_ + c * 32 + half * 16);
        acc = wmma_bf16(a, b, acc);
    }

    const int col  = col0 + l16;
    const float bc = bias[col];
    const int h = col >> 7, d = col & 127;
#pragma unroll
    for (int r = 0; r < 8; ++r) {
        const int row = row0 + r + half * 8;   // 0..4095
        const int bb = row >> 11, ss = row & (S_ - 1);
        float v = acc[r] + bc;
        if (MODE == 0) v *= mask[row] ? QSCALE : 0.0f;  // masked row => const scores
        if (MODE == 2)
            out[((size_t)(bb * H_ + h) * D_ + d) * S_ + ss] = (bf16)v;
        else
            out[((size_t)(bb * H_ + h) * S_ + ss) * D_ + d] = (bf16)v;
    }
}

// ---------------------------------------------------------------------------
// Flash attention. Block = 4 waves; each wave owns 16 query rows, d=128.
// K/V 64-key tiles are DMA'd into LDS by the TDM (wave 0 issues, double buffer),
// all 4 waves consume them: S = Q K^T (16 WMMA), online softmax, P@V (16 WMMA).
#define PPAD 72    // P tile: 64 keys padded to 72 bf16  (144B stride, no conflicts)
#define KPAD 136   // K tile row: 128 bf16 + 8 pad       (272B stride)
#define VPAD 72    // V tile row:  64 bf16 + 8 pad       (144B stride)
__global__ __launch_bounds__(128) void k_attn(const bf16* __restrict__ Q,
                                              const bf16* __restrict__ K,
                                              const bf16* __restrict__ VT,
                                              bf16* __restrict__ CTX) {
    __shared__ __align__(16) bf16 pbuf[4][16][PPAD];
    __shared__ __align__(16) bf16 kbuf[2][64][KPAD];
    __shared__ __align__(16) bf16 vbuf[2][128][VPAD];

    const int wave = threadIdx.x >> 5;
    const int lane = threadIdx.x & 31;
    const int half = lane >> 4;
    const int l16  = lane & 15;
    const int bh   = blockIdx.y;                  // b*8 + h
    const int q0   = blockIdx.x * 64 + wave * 16; // query row tile

    const bf16* Qb = Q  + (size_t)bh * S_ * D_;
    const bf16* Kb = K  + (size_t)bh * S_ * D_;
    const bf16* Vb = VT + (size_t)bh * D_ * S_;

    // Q fragments stay resident in VGPRs (already scaled by log2e/sqrt(d))
    bf16x16 qf[4];
    {
        const bf16* qrow = Qb + (size_t)(q0 + l16) * D_;
#pragma unroll
        for (int c = 0; c < 4; ++c) qf[c] = load_afrag(qrow, c * 32, half);
    }

    floatx8 acc[8];
#pragma unroll
    for (int j = 0; j < 8; ++j) acc[j] = (floatx8)(0.0f);
    float mrow[8], lrow[8];
#pragma unroll
    for (int r = 0; r < 8; ++r) { mrow[r] = -__builtin_inff(); lrow[r] = 0.0f; }

    // prologue: TDM prefetch of key-block 0 into buffer 0 (wave 0 only)
    if (wave == 0) {
        tdm_load_2d((u32)(uintptr_t)&kbuf[0][0][0], Kb, D_, S_, D_, 64, D_, TDM_FLAGS_K);
        tdm_load_2d((u32)(uintptr_t)&vbuf[0][0][0], Vb, S_, D_, 64, D_, S_, TDM_FLAGS_V);
    }

    for (int kb = 0; kb < NKB; ++kb) {
        const int cur = kb & 1;
        if (wave == 0) __builtin_amdgcn_s_wait_tensorcnt(0);
        __syncthreads();   // tile `cur` visible to all waves; `cur^1` free again
        if (wave == 0 && kb + 1 < NKB) {
            const int kn = (kb + 1) * 64;
            tdm_load_2d((u32)(uintptr_t)&kbuf[cur ^ 1][0][0],
                        Kb + (size_t)kn * D_, D_, S_, D_, 64, D_, TDM_FLAGS_K);
            tdm_load_2d((u32)(uintptr_t)&vbuf[cur ^ 1][0][0],
                        Vb + kn, S_, D_, 64, D_, S_, TDM_FLAGS_V);
        }

        // ---- scores: 4 column tiles of 16 keys, K-dim = d = 128 (from LDS)
        floatx8 s[4];
#pragma unroll
        for (int t = 0; t < 4; ++t) {
            floatx8 sv = (floatx8)(0.0f);
            const bf16* krow = &kbuf[cur][t * 16 + l16][0];
#pragma unroll
            for (int c = 0; c < 4; ++c) {
                bf16x16 kf = load_bfrag16(krow, c * 32 + half * 16);
                sv = wmma_bf16(qf[c], kf, sv);
            }
            s[t] = sv;
        }

        // ---- online softmax (rows live per v8f slot; reduce across 16 lanes)
        float mnew[8], alpha[8];
#pragma unroll
        for (int r = 0; r < 8; ++r) {
            float v = fmaxf(fmaxf(s[0][r], s[1][r]), fmaxf(s[2][r], s[3][r]));
#pragma unroll
            for (int off = 1; off < 16; off <<= 1)
                v = fmaxf(v, __shfl_xor(v, off, 32));
            mnew[r]  = fmaxf(mrow[r], v);
            alpha[r] = __builtin_amdgcn_exp2f(mrow[r] - mnew[r]);
            mrow[r]  = mnew[r];
        }
#pragma unroll
        for (int t = 0; t < 4; ++t)
#pragma unroll
            for (int r = 0; r < 8; ++r)
                s[t][r] = __builtin_amdgcn_exp2f(s[t][r] - mnew[r]);
#pragma unroll
        for (int r = 0; r < 8; ++r) {
            float v = (s[0][r] + s[1][r]) + (s[2][r] + s[3][r]);
#pragma unroll
            for (int off = 1; off < 16; off <<= 1)
                v += __shfl_xor(v, off, 32);
            lrow[r] = lrow[r] * alpha[r] + v;
        }
#pragma unroll
        for (int j = 0; j < 8; ++j)
#pragma unroll
            for (int r = 0; r < 8; ++r) acc[j][r] *= alpha[r];

        // ---- P: C-layout -> A-layout via per-wave LDS tile (bf16)
#pragma unroll
        for (int t = 0; t < 4; ++t)
#pragma unroll
            for (int r = 0; r < 8; ++r)
                pbuf[wave][r + half * 8][t * 16 + l16] = (bf16)s[t][r];

        bf16x16 pf[2];
#pragma unroll
        for (int c = 0; c < 2; ++c) {
            const bf16* prow = &pbuf[wave][l16][c * 32 + half * 16];
            pf[c] = join16(*(const bf16x8*)(prow), *(const bf16x8*)(prow + 8));
        }

        // ---- ctx += P(16x64) @ V(64x128); V rows (d) in LDS, keys contiguous
#pragma unroll
        for (int c = 0; c < 2; ++c)
#pragma unroll
            for (int j = 0; j < 8; ++j) {
                const bf16* vrow = &vbuf[cur][j * 16 + l16][0];
                bf16x16 vf = load_bfrag16(vrow, c * 32 + half * 16);
                acc[j] = wmma_bf16(pf[c], vf, acc[j]);
            }
    }

    // ---- epilogue: normalize, store ctx bf16 as [b, s, h*128+d]
    const int bb = bh >> 3, hh = bh & 7;
    float inv[8];
#pragma unroll
    for (int r = 0; r < 8; ++r) inv[r] = 1.0f / lrow[r];
#pragma unroll
    for (int j = 0; j < 8; ++j) {
        const int d = hh * D_ + j * 16 + l16;
#pragma unroll
        for (int r = 0; r < 8; ++r) {
            const int ss = q0 + r + half * 8;
            CTX[(size_t)(bb * S_ + ss) * DP_ + d] = (bf16)(acc[j][r] * inv[r]);
        }
    }
}

// ---------------------------------------------------------------------------
// Output GEMM: out[4096,128] = ctx[4096,1024](bf16) @ Wo (stored Wo^T) + bo
__global__ __launch_bounds__(256) void k_outproj(const bf16* __restrict__ CTX,
                                                 const bf16* __restrict__ WOT,
                                                 const float* __restrict__ bo,
                                                 float* __restrict__ out) {
    const int wave = threadIdx.x >> 5;
    const int lane = threadIdx.x & 31;
    const int half = lane >> 4;
    const int l16  = lane & 15;
    const int tile = blockIdx.x * 8 + wave;   // 256 * 8 tiles
    const int mt = tile >> 3;
    const int nt = tile & 7;
    const int row0 = mt * 16;
    const int col0 = nt * 16;

    floatx8 acc = (floatx8)(0.0f);
    const bf16* arow = CTX + (size_t)(row0 + l16) * DP_;
    const bf16* brow = WOT + (size_t)(col0 + l16) * DP_;
#pragma unroll 8
    for (int c = 0; c < 32; ++c) {
        bf16x16 a = load_afrag(arow, c * 32, half);
        bf16x16 b = *(const bf16x16*)(brow + c * 32 + half * 16);
        acc = wmma_bf16(a, b, acc);
    }

    const int col = col0 + l16;
    const float bc = bo[col];
#pragma unroll
    for (int r = 0; r < 8; ++r)
        out[(size_t)(row0 + r + half * 8) * D_ + col] = acc[r] + bc;
}

// ---------------------------------------------------------------------------
extern "C" void kernel_launch(void* const* d_in, const int* in_sizes, int n_in,
                              void* d_out, int out_size, void* d_ws, size_t ws_size,
                              hipStream_t stream) {
    (void)in_sizes; (void)n_in; (void)out_size; (void)ws_size;
    const float* query = (const float*)d_in[0];
    const float* key   = (const float*)d_in[1];
    const float* value = (const float*)d_in[2];
    const unsigned char* mask = (const unsigned char*)d_in[3];
    const float* Wq = (const float*)d_in[4];
    const float* bq = (const float*)d_in[5];
    const float* Wk = (const float*)d_in[6];
    const float* bk = (const float*)d_in[7];
    const float* Wv = (const float*)d_in[8];
    const float* bv = (const float*)d_in[9];
    const float* Wo = (const float*)d_in[10];
    const float* bo = (const float*)d_in[11];
    float* out = (float*)d_out;

    // workspace carve-up (~36 MB; stays L2-resident on MI455X's 192MB L2)
    char* ws = (char*)d_ws;
    size_t off = 0;
    auto carve = [&](size_t bytes) { char* p = ws + off; off += (bytes + 255) & ~(size_t)255; return p; };
    bf16* XQ  = (bf16*)carve((size_t)ROWS * D_ * 2);
    bf16* XK  = (bf16*)carve((size_t)ROWS * D_ * 2);
    bf16* XV  = (bf16*)carve((size_t)ROWS * D_ * 2);
    bf16* WQT = (bf16*)carve((size_t)DP_ * D_ * 2);
    bf16* WKT = (bf16*)carve((size_t)DP_ * D_ * 2);
    bf16* WVT = (bf16*)carve((size_t)DP_ * D_ * 2);
    bf16* WOT = (bf16*)carve((size_t)D_ * DP_ * 2);
    bf16* Qw  = (bf16*)carve((size_t)B_ * H_ * S_ * D_ * 2);
    bf16* Kw  = (bf16*)carve((size_t)B_ * H_ * S_ * D_ * 2);
    bf16* Vw  = (bf16*)carve((size_t)B_ * H_ * S_ * D_ * 2);
    bf16* CTX = (bf16*)carve((size_t)ROWS * DP_ * 2);

    const int nX = ROWS * D_;   // 524288
    const int nW = D_ * DP_;    // 131072

    k_cvt_bf16<<<nX / 256, 256, 0, stream>>>(query, XQ, nX);
    k_cvt_bf16<<<nX / 256, 256, 0, stream>>>(key,   XK, nX);
    k_cvt_bf16<<<nX / 256, 256, 0, stream>>>(value, XV, nX);
    k_cvtT_bf16<<<nW / 256, 256, 0, stream>>>(Wq, WQT, D_, DP_);
    k_cvtT_bf16<<<nW / 256, 256, 0, stream>>>(Wk, WKT, D_, DP_);
    k_cvtT_bf16<<<nW / 256, 256, 0, stream>>>(Wv, WVT, D_, DP_);
    k_cvtT_bf16<<<nW / 256, 256, 0, stream>>>(Wo, WOT, DP_, D_);

    k_proj<0><<<2048, 256, 0, stream>>>(XQ, WQT, bq, mask, Qw);
    k_proj<1><<<2048, 256, 0, stream>>>(XK, WKT, bk, mask, Kw);
    k_proj<2><<<2048, 256, 0, stream>>>(XV, WVT, bv, mask, Vw);

    k_attn<<<dim3(S_ / 64, B_ * H_), 128, 0, stream>>>(Qw, Kw, Vw, CTX);

    k_outproj<<<256, 256, 0, stream>>>(CTX, WOT, bo, out);
}
